// SwinBlock_25486335934874
// MI455X (gfx1250) — compile-verified, matrix operand order checked by reading, not run on hardware
//
#include <hip/hip_runtime.h>
#include <hip/hip_bf16.h>

// ---------------------------------------------------------------------------
// Swin block on gfx1250 (CDNA5): all matmuls via v_wmma_f32_16x16x32_f16.
// GEMM waves are register-blocked 32x64 (8 WMMAs per 12 b128 loads).
// ---------------------------------------------------------------------------

typedef __attribute__((ext_vector_type(16))) _Float16 v16h;
typedef __attribute__((ext_vector_type(8)))  float    v8f;

#define DIM      256
#define HEADS    8
#define HEAD_DIM 32
#define WIN      8
#define SHIFT    4
#define IMG      64          // H == W
#define NTOK     64          // tokens per window
#define NW       64          // windows per image
#define BATCH    32
#define TOK      (BATCH * IMG * IMG)   // 131072 tokens total
#define HIDDEN   1024
#define SCALE_QK 0.17677669529663687f  // 32^-0.5
#define LN_EPS   1e-5f

// ---------------------------------------------------------------------------
// WMMA 16-bit A/B fragment loader.
// ISA layout (16-bit A 16x32): lane half = lane>>4.
//   elements e=0..7  hold K = kbase + 8*half + e        (16 contiguous bytes)
//   elements e=8..15 hold K = kbase + 16 + 8*half + e-8 (16 contiguous bytes)
// With row-major (stride-K) storage both chunks are single b128 loads.
// B fragments use the mirrored layout (column per lane), so storing B
// transposed (N x K) makes its loads identical.
// ---------------------------------------------------------------------------
template <typename P>
__device__ __forceinline__ v16h load_frag(const P* row, int kbase, int half) {
    union { uint4 u[2]; v16h v; } cvt;
    cvt.u[0] = *(const uint4*)(row + kbase + 8 * half);
    cvt.u[1] = *(const uint4*)(row + kbase + 16 + 8 * half);
    return cvt.v;
}

__device__ __forceinline__ v8f wmma_f16(v16h a, v16h b, v8f c) {
    return __builtin_amdgcn_wmma_f32_16x16x32_f16(false, a, false, b,
                                                  (short)0, c, false, false);
}

// ---------------------------------------------------------------------------
// Weight prep: f32 (K x N) -> f16 transposed (N x K)
// ---------------------------------------------------------------------------
__global__ void transpose_f16_kernel(const float* __restrict__ W,
                                     _Float16* __restrict__ Wt, int K, int N) {
    int i = blockIdx.x * 256 + threadIdx.x;
    if (i >= K * N) return;
    int k = i / N, n = i % N;
    Wt[(size_t)n * K + k] = (_Float16)W[i];
}

// ---------------------------------------------------------------------------
// LayerNorm kernels. One block = one token (256 channels, 8 waves).
// mode SHIFTED: also applies roll(-4,-4) + window partition into f16 buffer.
// ---------------------------------------------------------------------------
template <int SHIFTED>
__global__ __launch_bounds__(256) void ln_kernel(const float* __restrict__ x,
                                                 const float* __restrict__ g,
                                                 const float* __restrict__ b,
                                                 _Float16* __restrict__ out) {
    int t  = blockIdx.x;          // token (image space, row-major)
    int ch = threadIdx.x;
    float v  = x[(size_t)t * DIM + ch];
    float s  = v, s2 = v * v;
    #pragma unroll
    for (int m = 16; m >= 1; m >>= 1) {
        s  += __shfl_xor(s,  m);
        s2 += __shfl_xor(s2, m);
    }
    __shared__ float ps[8], ps2[8];
    int wid = threadIdx.x >> 5;
    if ((threadIdx.x & 31) == 0) { ps[wid] = s; ps2[wid] = s2; }
    __syncthreads();
    s = 0.f; s2 = 0.f;
    #pragma unroll
    for (int i = 0; i < 8; i++) { s += ps[i]; s2 += ps2[i]; }
    float mu  = s * (1.0f / DIM);
    float var = s2 * (1.0f / DIM) - mu * mu;
    float y = (v - mu) * rsqrtf(var + LN_EPS) * g[ch] + b[ch];

    size_t dst;
    if (SHIFTED) {
        int bi = t >> 12, pix = t & 4095;
        int hh = pix >> 6, ww = pix & 63;
        int hr = (hh + IMG - SHIFT) & 63;      // roll by -SHIFT
        int wr = (ww + IMG - SHIFT) & 63;
        int w  = bi * NW + (hr >> 3) * 8 + (wr >> 3);
        int n  = (hr & 7) * 8 + (wr & 7);
        dst = ((size_t)w * NTOK + n) * DIM + ch;
    } else {
        dst = (size_t)t * DIM + ch;
    }
    out[dst] = (_Float16)y;
}

// ---------------------------------------------------------------------------
// Register-blocked WMMA GEMM: C[M,N] = A[M,K] * Bt[N,K]^T + bias.
// One wave computes a 32x64 output block (2 M-tiles x 4 N-tiles):
// per 32-deep k-step: 2 A frags + 4 B frags (12 b128 loads) -> 8 WMMAs.
// MODE 0: -> f16 Q/K layout (w, head, tok, d)
// MODE 1: -> f16 V  layout (w, head, d, tok)       (transposed for P*V frag)
// MODE 2: proj: window-reverse + roll(+4,+4) + residual -> f32 image space
// MODE 3: fc1 + exact GELU -> f16 row-major
// MODE 4: fc2 + residual   -> f32 row-major (final output)
// ---------------------------------------------------------------------------
template <int KDIM, int MODE>
__global__ __launch_bounds__(256) void gemm_wmma_kernel(
        const _Float16* __restrict__ A, const _Float16* __restrict__ Bt,
        const float* __restrict__ bias, void* __restrict__ outp,
        const float* __restrict__ resid, int Mgrp, int Ngrp, int Nout) {
    int gw   = (blockIdx.x * blockDim.x + threadIdx.x) >> 5;
    int lane = threadIdx.x & 31;
    int gm   = gw / Ngrp, gn = gw % Ngrp;      // 32-row / 64-col groups
    if (gm >= Mgrp) return;                    // uniform per wave -> EXEC full
    int half = lane >> 4, lc = lane & 15;

    const _Float16* arow0 = A + (size_t)(gm * 32 + lc) * KDIM;
    const _Float16* arow1 = A + (size_t)(gm * 32 + 16 + lc) * KDIM;
    const _Float16* brow[4];
    #pragma unroll
    for (int j = 0; j < 4; j++)
        brow[j] = Bt + (size_t)(gn * 64 + j * 16 + lc) * KDIM;

    v8f acc[2][4];
    #pragma unroll
    for (int i = 0; i < 2; i++)
        #pragma unroll
        for (int j = 0; j < 4; j++) acc[i][j] = (v8f){};

    #pragma unroll 2
    for (int k = 0; k < KDIM; k += 32) {
        if (k + 32 < KDIM) {                       // gfx1250 global_prefetch_b8
            __builtin_prefetch(arow0 + k + 32, 0, 3);
            __builtin_prefetch(arow1 + k + 32, 0, 3);
        }
        v16h a0 = load_frag(arow0, k, half);
        v16h a1 = load_frag(arow1, k, half);
        #pragma unroll
        for (int j = 0; j < 4; j++) {
            v16h b = load_frag(brow[j], k, half);
            acc[0][j] = wmma_f16(a0, b, acc[0][j]);
            acc[1][j] = wmma_f16(a1, b, acc[1][j]);
        }
    }

    #pragma unroll
    for (int i = 0; i < 2; i++) {
        #pragma unroll
        for (int j = 0; j < 4; j++) {
            int n  = gn * 64 + j * 16 + lc;
            float bn = bias[n];
            #pragma unroll
            for (int r = 0; r < 8; r++) {
                int m = gm * 32 + i * 16 + half * 8 + r;   // token / row index
                float val = acc[i][j][r] + bn;
                if (MODE == 0) {                            // Q or K
                    int w = m >> 6, tok = m & 63, h = n >> 5, d = n & 31;
                    ((_Float16*)outp)[(((size_t)(w * HEADS + h) * NTOK + tok) << 5) + d] =
                        (_Float16)val;
                } else if (MODE == 1) {                     // V transposed
                    int w = m >> 6, tok = m & 63, h = n >> 5, d = n & 31;
                    ((_Float16*)outp)[(((size_t)(w * HEADS + h) * HEAD_DIM + d) << 6) + tok] =
                        (_Float16)val;
                } else if (MODE == 2) {                     // proj scatter + residual
                    int w = m >> 6, nn = m & 63;
                    int bi = w >> 6, wl = w & 63;
                    int hr = (wl >> 3) * 8 + (nn >> 3);
                    int wr = (wl & 7) * 8 + (nn & 7);
                    int hh = (hr + SHIFT) & 63;             // roll back by +SHIFT
                    int ww = (wr + SHIFT) & 63;
                    size_t idx = (((size_t)bi << 12) + hh * 64 + ww) * DIM + n;
                    ((float*)outp)[idx] = resid[idx] + val;
                } else if (MODE == 3) {                     // fc1 + exact GELU
                    float gel = 0.5f * val * (1.0f + erff(val * 0.70710678118f));
                    ((_Float16*)outp)[(size_t)m * Nout + n] = (_Float16)gel;
                } else {                                    // fc2 + residual
                    size_t idx = (size_t)m * Nout + n;
                    ((float*)outp)[idx] = resid[idx] + val;
                }
            }
        }
    }
}

// ---------------------------------------------------------------------------
// Attention: one workgroup per window (8 waves = 8 heads).
//   S = Q K^T * scale + rel_bias + mask ; softmax ; O = P V
// P goes through LDS (f16) to convert C-layout -> A-fragment layout.
// ---------------------------------------------------------------------------
__global__ __launch_bounds__(256) void attn_kernel(
        const _Float16* __restrict__ q, const _Float16* __restrict__ k,
        const _Float16* __restrict__ vt, const float* __restrict__ rel_bias,
        const int* __restrict__ pos_index, const float* __restrict__ attn_mask,
        _Float16* __restrict__ attn_out) {
    __shared__ _Float16 pbuf[HEADS * NTOK * NTOK];  // 64 KB, per-wave private slice

    int lane = threadIdx.x & 31;
    int h    = threadIdx.x >> 5;        // head = wave id
    int w    = blockIdx.x;              // global window
    int wl   = w & 63;                  // window within image (mask index)
    int half = lane >> 4, lc = lane & 15;

    const _Float16* qb = q  + (size_t)(w * HEADS + h) * (NTOK * HEAD_DIM);
    const _Float16* kb = k  + (size_t)(w * HEADS + h) * (NTOK * HEAD_DIM);
    const _Float16* vb = vt + (size_t)(w * HEADS + h) * (NTOK * HEAD_DIM);

    // ---- S = Q K^T : 4x4 tiles of 16x16, K-dim = 32 (single WMMA each) ----
    v16h af[4], bf[4];
    #pragma unroll
    for (int t = 0; t < 4; t++) {
        af[t] = load_frag(qb + (t * 16 + lc) * HEAD_DIM, 0, half);
        bf[t] = load_frag(kb + (t * 16 + lc) * HEAD_DIM, 0, half);
    }
    v8f s[4][4];
    #pragma unroll
    for (int ti = 0; ti < 4; ti++)
        #pragma unroll
        for (int tj = 0; tj < 4; tj++) {
            v8f z = {};
            s[ti][tj] = wmma_f16(af[ti], bf[tj], z);
        }

    // ---- scale + rel_bias + mask + row softmax (rows live in lane groups of 16)
    _Float16* prow = pbuf + h * (NTOK * NTOK);
    #pragma unroll
    for (int ti = 0; ti < 4; ti++) {
        #pragma unroll
        for (int r = 0; r < 8; r++) {
            int m = ti * 16 + half * 8 + r;
            float vals[4];
            #pragma unroll
            for (int tj = 0; tj < 4; tj++) {
                int n = tj * 16 + lc;
                vals[tj] = s[ti][tj][r] * SCALE_QK
                         + rel_bias[pos_index[m * NTOK + n] * HEADS + h]
                         + attn_mask[((size_t)wl * NTOK + m) * NTOK + n];
            }
            float mx = fmaxf(fmaxf(vals[0], vals[1]), fmaxf(vals[2], vals[3]));
            #pragma unroll
            for (int msk = 8; msk >= 1; msk >>= 1)
                mx = fmaxf(mx, __shfl_xor(mx, msk));
            float sum = 0.f;
            #pragma unroll
            for (int tj = 0; tj < 4; tj++) {
                vals[tj] = __expf(vals[tj] - mx);
                sum += vals[tj];
            }
            #pragma unroll
            for (int msk = 8; msk >= 1; msk >>= 1)
                sum += __shfl_xor(sum, msk);
            float inv = 1.0f / sum;
            #pragma unroll
            for (int tj = 0; tj < 4; tj++)
                prow[m * NTOK + tj * 16 + lc] = (_Float16)(vals[tj] * inv);
        }
    }
    // each wave only reads its own LDS slice; dscnt ordering handled by compiler

    // ---- O = P V : 4 row-tiles x 2 d-tiles, K-dim = 64 (two WMMA steps) ----
    #pragma unroll
    for (int ti = 0; ti < 4; ti++) {
        v16h pa0 = load_frag(prow + (ti * 16 + lc) * NTOK, 0,  half);
        v16h pa1 = load_frag(prow + (ti * 16 + lc) * NTOK, 32, half);
        #pragma unroll
        for (int tj = 0; tj < 2; tj++) {
            v16h vb0 = load_frag(vb + (tj * 16 + lc) * NTOK, 0,  half);
            v16h vb1 = load_frag(vb + (tj * 16 + lc) * NTOK, 32, half);
            v8f o = {};
            o = wmma_f16(pa0, vb0, o);
            o = wmma_f16(pa1, vb1, o);
            #pragma unroll
            for (int r = 0; r < 8; r++) {
                int m = ti * 16 + half * 8 + r;
                int d = tj * 16 + lc;
                attn_out[((size_t)w * NTOK + m) * DIM + h * HEAD_DIM + d] =
                    (_Float16)o[r];
            }
        }
    }
}

// ---------------------------------------------------------------------------
// Host-side launch
// ---------------------------------------------------------------------------
extern "C" void kernel_launch(void* const* d_in, const int* in_sizes, int n_in,
                              void* d_out, int out_size, void* d_ws, size_t ws_size,
                              hipStream_t stream) {
    const float* x      = (const float*)d_in[0];
    const float* ln1_g  = (const float*)d_in[1];
    const float* ln1_b  = (const float*)d_in[2];
    const float* wq     = (const float*)d_in[3];
    const float* bq     = (const float*)d_in[4];
    const float* wk     = (const float*)d_in[5];
    const float* bk     = (const float*)d_in[6];
    const float* wv     = (const float*)d_in[7];
    const float* bv     = (const float*)d_in[8];
    const float* wp     = (const float*)d_in[9];
    const float* bp     = (const float*)d_in[10];
    const float* relb   = (const float*)d_in[11];
    const float* ln2_g  = (const float*)d_in[12];
    const float* ln2_b  = (const float*)d_in[13];
    const float* w1     = (const float*)d_in[14];
    const float* b1     = (const float*)d_in[15];
    const float* w2     = (const float*)d_in[16];
    const float* b2     = (const float*)d_in[17];
    const int*   posi   = (const int*)d_in[18];
    const float* amask  = (const float*)d_in[19];
    float*       out    = (float*)d_out;

    char* ws = (char*)d_ws;
    size_t o = 0;
    auto alloc = [&](size_t bytes) { char* p = ws + o; o += (bytes + 255) & ~(size_t)255; return p; };

    _Float16* wqt  = (_Float16*)alloc((size_t)DIM * DIM * 2);
    _Float16* wkt  = (_Float16*)alloc((size_t)DIM * DIM * 2);
    _Float16* wvt  = (_Float16*)alloc((size_t)DIM * DIM * 2);
    _Float16* wpt  = (_Float16*)alloc((size_t)DIM * DIM * 2);
    _Float16* w1t  = (_Float16*)alloc((size_t)DIM * HIDDEN * 2);
    _Float16* w2t  = (_Float16*)alloc((size_t)HIDDEN * DIM * 2);
    _Float16* wins = (_Float16*)alloc((size_t)TOK * DIM * 2);
    _Float16* qb   = (_Float16*)alloc((size_t)TOK * DIM * 2);
    _Float16* kb   = (_Float16*)alloc((size_t)TOK * DIM * 2);
    _Float16* vtb  = (_Float16*)alloc((size_t)TOK * DIM * 2);
    _Float16* atto = (_Float16*)alloc((size_t)TOK * DIM * 2);
    float*    xbuf = (float*)   alloc((size_t)TOK * DIM * 4);
    _Float16* ln2o = (_Float16*)alloc((size_t)TOK * DIM * 2);
    _Float16* hid  = (_Float16*)alloc((size_t)TOK * HIDDEN * 2);
    (void)ws_size; (void)in_sizes; (void)n_in; (void)out_size;

    // 1) weight prep (f32 KxN -> f16 NxK)
    transpose_f16_kernel<<<(DIM * DIM + 255) / 256, 256, 0, stream>>>(wq, wqt, DIM, DIM);
    transpose_f16_kernel<<<(DIM * DIM + 255) / 256, 256, 0, stream>>>(wk, wkt, DIM, DIM);
    transpose_f16_kernel<<<(DIM * DIM + 255) / 256, 256, 0, stream>>>(wv, wvt, DIM, DIM);
    transpose_f16_kernel<<<(DIM * DIM + 255) / 256, 256, 0, stream>>>(wp, wpt, DIM, DIM);
    transpose_f16_kernel<<<(DIM * HIDDEN + 255) / 256, 256, 0, stream>>>(w1, w1t, DIM, HIDDEN);
    transpose_f16_kernel<<<(HIDDEN * DIM + 255) / 256, 256, 0, stream>>>(w2, w2t, HIDDEN, DIM);

    // 2) LN1 + roll(-4,-4) + window partition -> f16
    ln_kernel<1><<<TOK, 256, 0, stream>>>(x, ln1_g, ln1_b, wins);

    // 3) Q, K, V projections (WMMA, 32x64 wave blocks)
    const int mg = TOK / 32;                          // 4096 M-groups
    const int ng256  = DIM / 64;                      // 4 N-groups
    const int ng1024 = HIDDEN / 64;                   // 16 N-groups
    const int blk256  = (mg * ng256 + 7) / 8;         // 8 waves / block
    const int blk1024 = (mg * ng1024 + 7) / 8;
    gemm_wmma_kernel<DIM, 0><<<blk256, 256, 0, stream>>>(wins, wqt, bq, qb,  nullptr, mg, ng256, DIM);
    gemm_wmma_kernel<DIM, 0><<<blk256, 256, 0, stream>>>(wins, wkt, bk, kb,  nullptr, mg, ng256, DIM);
    gemm_wmma_kernel<DIM, 1><<<blk256, 256, 0, stream>>>(wins, wvt, bv, vtb, nullptr, mg, ng256, DIM);

    // 4) windowed attention (WMMA S + softmax + WMMA PV)
    attn_kernel<<<TOK / NTOK, 256, 0, stream>>>(qb, kb, vtb, relb, posi, amask, atto);

    // 5) output projection + window-reverse + roll(+4,+4) + residual
    gemm_wmma_kernel<DIM, 2><<<blk256, 256, 0, stream>>>(atto, wpt, bp, xbuf, x, mg, ng256, DIM);

    // 6) LN2 -> f16
    ln_kernel<0><<<TOK, 256, 0, stream>>>(xbuf, ln2_g, ln2_b, ln2o);

    // 7) MLP fc1 + exact GELU
    gemm_wmma_kernel<DIM, 3><<<blk1024, 256, 0, stream>>>(ln2o, w1t, b1, hid, nullptr, mg, ng1024, HIDDEN);

    // 8) MLP fc2 + residual -> final f32 output
    gemm_wmma_kernel<HIDDEN, 4><<<blk256, 256, 0, stream>>>(hid, w2t, b2, out, xbuf, mg, ng256, DIM);
}
